// GptLanguageModel_23373212025031
// MI455X (gfx1250) — matrix operand order repaired and checked by reading, not compiled, not run on hardware
//
#include <hip/hip_runtime.h>
#include <hip/hip_bf16.h>

// ---- problem constants (match reference) ----
#define D_    1024
#define T_    1024
#define B_    4
#define H_    16
#define HD_   64
#define L_    6
#define NTOT  (B_*T_*D_)      // 4,194,304 elements in [B,T,D]
#define MROWS (B_*T_)         // 4096
#define NPART 1024

// ---- CDNA5 WMMA types (per CDNA5_HIP.md / ISA 7.12) ----
typedef __attribute__((ext_vector_type(16))) __bf16        v16bf;
typedef __attribute__((ext_vector_type(8)))  float         v8f;
typedef __attribute__((ext_vector_type(4)))  unsigned int  v4u;

union FragAB { v16bf v; v4u q[2]; };

static __device__ __forceinline__ unsigned short f2bf(float f) {
  unsigned u = __float_as_uint(f);
  unsigned r = u + 0x7FFFu + ((u >> 16) & 1u);   // round-to-nearest-even
  return (unsigned short)(r >> 16);
}

static __device__ __forceinline__ v8f wmma_bf16(v16bf a, v16bf b, v8f c) {
  // D = A(16x32 bf16) x B(32x16 bf16) + C(16x16 f32)
  return __builtin_amdgcn_wmma_f32_16x16x32_bf16(
      /*neg_a=*/false, a, /*neg_b=*/false, b,
      /*c_mod=*/(short)0, c, /*reuse_a=*/false, /*reuse_b=*/false);
}

// ============================================================
// Embedding: x[b,t,d] = tok[idx[b,t],d] + pos[t,d]
// ============================================================
__global__ void k_embed(const int* __restrict__ idx, const float* __restrict__ tok,
                        const float* __restrict__ pos, float* __restrict__ x) {
  int i = blockIdx.x * 256 + threadIdx.x;       // 0..NTOT-1
  int row = i >> 10;                            // (b*T + t)
  int d   = i & (D_ - 1);
  int t   = row & (T_ - 1);
  x[i] = tok[(long)idx[row] * D_ + d] + pos[t * D_ + d];
}

// ============================================================
// Weight fp32 [K][N] -> bf16 transposed [N][K]
// ============================================================
__global__ void k_transpose_bf16(const float* __restrict__ W,
                                 unsigned short* __restrict__ WT, int K, int N) {
  int i = blockIdx.x * 256 + threadIdx.x;       // K*N threads
  int k = i / N, n = i % N;
  WT[(long)n * K + k] = f2bf(W[i]);
}

// ============================================================
// Whole-tensor LN statistics (deterministic two-stage)
// ============================================================
__global__ void k_reduce(const float* __restrict__ a, const float* __restrict__ b,
                         float* __restrict__ part, int n) {
  __shared__ float s1[256], s2[256];
  float s = 0.f, ss = 0.f;
  for (int i = blockIdx.x * 256 + threadIdx.x; i < n; i += gridDim.x * 256) {
    float v = a[i];
    if (b) v += b[i];
    s += v; ss += v * v;
  }
  s1[threadIdx.x] = s; s2[threadIdx.x] = ss;
  __syncthreads();
  for (int o = 128; o > 0; o >>= 1) {
    if (threadIdx.x < o) {
      s1[threadIdx.x] += s1[threadIdx.x + o];
      s2[threadIdx.x] += s2[threadIdx.x + o];
    }
    __syncthreads();
  }
  if (threadIdx.x == 0) {
    part[2 * blockIdx.x]     = s1[0];
    part[2 * blockIdx.x + 1] = s2[0];
  }
}

__global__ void k_finalize(const float* __restrict__ part, float* __restrict__ acc) {
  __shared__ float s1[256], s2[256];
  float s = 0.f, ss = 0.f;
  for (int i = threadIdx.x; i < NPART; i += 256) {
    s += part[2 * i]; ss += part[2 * i + 1];
  }
  s1[threadIdx.x] = s; s2[threadIdx.x] = ss;
  __syncthreads();
  for (int o = 128; o > 0; o >>= 1) {
    if (threadIdx.x < o) {
      s1[threadIdx.x] += s1[threadIdx.x + o];
      s2[threadIdx.x] += s2[threadIdx.x + o];
    }
    __syncthreads();
  }
  if (threadIdx.x == 0) { acc[0] = s1[0]; acc[1] = s2[0]; }
}

static __device__ __forceinline__ void ln_params(const float* acc, float& mu, float& rs) {
  const float inv_n = 1.f / (float)NTOT;
  mu = acc[0] * inv_n;
  float var = acc[1] * inv_n - mu * mu;
  rs = rsqrtf(var + 1e-5f);
}

// xn = ln(x); also bf16 copy for WMMA
__global__ void k_ln_xn(const float* __restrict__ x, const float* __restrict__ acc,
                        float* __restrict__ xn, unsigned short* __restrict__ xnb) {
  int i = blockIdx.x * 256 + threadIdx.x;
  float mu, rs; ln_params(acc, mu, rs);
  float v = (x[i] - mu) * rs;
  xn[i] = v;
  xnb[i] = f2bf(v);
}

// x = 2*( ln(xn+out) + out )   (stats taken over xn+out)
__global__ void k_ln_res(const float* __restrict__ xn, const float* __restrict__ po,
                         const float* __restrict__ acc, float* __restrict__ x) {
  int i = blockIdx.x * 256 + threadIdx.x;
  float mu, rs; ln_params(acc, mu, rs);
  float t = xn[i] + po[i];
  float v = (t - mu) * rs + po[i];
  x[i] = v + v;
}

__global__ void k_ln_final(const float* __restrict__ x, const float* __restrict__ acc,
                           float* __restrict__ dst) {
  int i = blockIdx.x * 256 + threadIdx.x;
  float mu, rs; ln_params(acc, mu, rs);
  dst[i] = (x[i] - mu) * rs;
}

// ============================================================
// GEMM: C[M,1024] = A[M,1024] * B, A bf16 [M][1024], B transposed bf16 [1024][1024].
// One wave computes a 32(M) x 64(N) tile: 8 f32 WMMA accumulators; each K-stage
// loads 2 A-fragments + 4 B-fragments and fires 8 WMMAs (B reused twice).
// Software-pipelined K loop (2 stages, K-step 64). __launch_bounds__(32,1)
// gives the allocator the full per-wave register file (no spills).
// MODE 0: f32 out; MODE 1: bf16 out; MODE 2: bf16 out transposed per-head [b,h][d][t].
// ============================================================
template <int MODE>
__global__ void __launch_bounds__(32, 1)
k_gemm(const unsigned short* __restrict__ A,
       const unsigned short* __restrict__ BT,
       float* __restrict__ Cf, unsigned short* __restrict__ Cb) {
  const int lane = threadIdx.x;
  const int half = lane >> 4, lm = lane & 15;
  const int mb = blockIdx.x * 32;
  const int nb = blockIdx.y * 64;

  const unsigned aoffU = (unsigned)((mb + lm) * D_ + half * 8);
  const unsigned aoffL = aoffU + 16 * D_;
  const unsigned boff  = (unsigned)((nb + lm) * D_ + half * 8);

  v8f acc[2][4] = {};
  FragAB aU0, aL0, aU1, aL1, b0[4], b1[4];

  auto loadAs = [&](FragAB& u, FragAB& l2, int kb) {
    u.q[0]  = *(const v4u*)(A + aoffU + kb);        // K = half*8 + 0..7
    u.q[1]  = *(const v4u*)(A + aoffU + kb + 16);   // K = 16 + half*8 + 0..7
    l2.q[0] = *(const v4u*)(A + aoffL + kb);
    l2.q[1] = *(const v4u*)(A + aoffL + kb + 16);
  };
  auto loadB = [&](FragAB* f, int kb) {
#pragma unroll
    for (int nt = 0; nt < 4; nt++) {
      f[nt].q[0] = *(const v4u*)(BT + boff + nt * (16 * D_) + kb);
      f[nt].q[1] = *(const v4u*)(BT + boff + nt * (16 * D_) + kb + 16);
    }
  };

  loadAs(aU0, aL0, 0);
  loadB(b0, 0);
  for (int kb = 0; kb < D_; kb += 64) {
    // stage 1 loads in flight while stage 0 WMMAs run
    loadAs(aU1, aL1, kb + 32);
    loadB(b1, kb + 32);
    __builtin_prefetch(A + aoffU + kb + 128, 0, 1);   // global_prefetch_b8
#pragma unroll
    for (int nt = 0; nt < 4; nt++) {
      acc[0][nt] = wmma_bf16(aU0.v, b0[nt].v, acc[0][nt]);
      acc[1][nt] = wmma_bf16(aL0.v, b0[nt].v, acc[1][nt]);
    }
    if (kb + 64 < D_) {
      loadAs(aU0, aL0, kb + 64);
      loadB(b0, kb + 64);
    }
#pragma unroll
    for (int nt = 0; nt < 4; nt++) {
      acc[0][nt] = wmma_bf16(aU1.v, b1[nt].v, acc[0][nt]);
      acc[1][nt] = wmma_bf16(aL1.v, b1[nt].v, acc[1][nt]);
    }
  }

#pragma unroll
  for (int g = 0; g < 2; g++) {
#pragma unroll
    for (int nt = 0; nt < 4; nt++) {
#pragma unroll
      for (int r = 0; r < 8; r++) {
        int m = mb + g * 16 + r + half * 8;
        int n = nb + nt * 16 + lm;
        if (MODE == 0) {
          Cf[(unsigned)(m * D_ + n)] = acc[g][nt][r];
        } else if (MODE == 1) {
          Cb[(unsigned)(m * D_ + n)] = f2bf(acc[g][nt][r]);
        } else {
          int b = m >> 10, t = m & (T_ - 1);
          int h = n >> 6,  d = n & (HD_ - 1);
          Cb[(unsigned)((((b * H_ + h) * HD_ + d) << 10) + t)] = f2bf(acc[g][nt][r]);
        }
      }
    }
  }
}

// ============================================================
// Attention (transposed-label einsum of the reference):
//   S[q,k] = (1/8) * sum_d kmat[b,h,q,d] * qmat[b,h,k,d], causal mask k<=q,
//   softmax over k, O[q,d] = sum_k P[q,k] * v[b,h,k,d].
// One wave per (b,h, 16-row q tile). Online softmax with batched cross-lane
// reductions. P staged in LDS bf16 16x32 and reloaded as an A-fragment for
// the P*V WMMAs. __launch_bounds__(32,1) keeps the loop-invariant A
// fragments resident (no scratch spills).
// q,k stored [B*T][H*HD] bf16; v stored transposed [b,h][d][t] bf16.
// ============================================================
__global__ void __launch_bounds__(32, 1)
k_attn(const unsigned short* __restrict__ q,
       const unsigned short* __restrict__ k,
       const unsigned short* __restrict__ vT,
       unsigned short* __restrict__ o) {
  __shared__ __align__(16) unsigned short pst[16 * 32];
  const int lane = threadIdx.x, half = lane >> 4, lm = lane & 15;
  const int qt = blockIdx.x;            // q tile index, 0..T/16-1
  const int bh = blockIdx.y;            // b*H + h
  const int b = bh >> 4, h = bh & 15;
  const int qb = qt * 16;

  // A fragments: rows = q indices of this tile, taken from the K tensor.
  const unsigned kaoff = (unsigned)((b * T_ + qb + lm) * D_ + h * HD_ + half * 16);
  FragAB A0, A1;
  A0.q[0] = *(const v4u*)(k + kaoff);
  A0.q[1] = *(const v4u*)(k + kaoff + 8);
  A1.q[0] = *(const v4u*)(k + kaoff + 32);
  A1.q[1] = *(const v4u*)(k + kaoff + 40);

  const unsigned qoff0 = (unsigned)((b * T_ + lm) * D_ + h * HD_ + half * 16);
  const unsigned voff0 = (unsigned)((((b * H_ + h) * HD_) << 10) + (lm << 10) + half * 16);

  float m[8], l[8];
  v8f O0 = {}, O1 = {}, O2 = {}, O3 = {};
#pragma unroll
  for (int r = 0; r < 8; r++) { m[r] = -__builtin_inff(); l[r] = 0.f; }

  for (int j = 0; j <= qt; j += 2) {     // process k-tiles in pairs (K=32 for P*V)
    // ---- preload ALL q B-fragments for this pair, then fire 4 WMMAs ----
    const unsigned q0 = qoff0 + ((unsigned)j << 14);       // + j*16*D_
    const unsigned q1 = q0 + (1u << 14);
    FragAB B00, B01, B10, B11;
    B00.q[0] = *(const v4u*)(q + q0);      B00.q[1] = *(const v4u*)(q + q0 + 8);
    B01.q[0] = *(const v4u*)(q + q0 + 32); B01.q[1] = *(const v4u*)(q + q0 + 40);
    B10.q[0] = *(const v4u*)(q + q1);      B10.q[1] = *(const v4u*)(q + q1 + 8);
    B11.q[0] = *(const v4u*)(q + q1 + 32); B11.q[1] = *(const v4u*)(q + q1 + 40);
    __builtin_prefetch(q + q0 + (2u << 14), 0, 1);

    v8f S0 = {}, S1 = {};
    S0 = wmma_bf16(A0.v, B00.v, S0);
    S0 = wmma_bf16(A1.v, B01.v, S0);
    S1 = wmma_bf16(A0.v, B10.v, S1);
    S1 = wmma_bf16(A1.v, B11.v, S1);

    // ---- issue V fragment loads now; softmax VALU below hides their latency
    const unsigned vo = voff0 + (unsigned)(j * 16);
    FragAB V0, V1, V2, V3;
    V0.q[0] = *(const v4u*)(vT + vo);              V0.q[1] = *(const v4u*)(vT + vo + 8);
    V1.q[0] = *(const v4u*)(vT + vo + (1u << 14)); V1.q[1] = *(const v4u*)(vT + vo + (1u << 14) + 8);
    V2.q[0] = *(const v4u*)(vT + vo + (2u << 14)); V2.q[1] = *(const v4u*)(vT + vo + (2u << 14) + 8);
    V3.q[0] = *(const v4u*)(vT + vo + (3u << 14)); V3.q[1] = *(const v4u*)(vT + vo + (3u << 14) + 8);

    // ---- scale + causal mask ----
    float e0[8], e1[8], rm[8];
#pragma unroll
    for (int r = 0; r < 8; r++) {
      int row = qb + r + half * 8;
      int c0 = j * 16 + lm;
      e0[r] = (c0      <= row) ? S0[r] * 0.125f : -__builtin_inff();
      e1[r] = (c0 + 16 <= row) ? S1[r] * 0.125f : -__builtin_inff();
      rm[r] = fmaxf(e0[r], e1[r]);
    }
    // ---- batched row-max reduction over the 16 lanes of each half ----
#pragma unroll
    for (int st = 1; st <= 8; st <<= 1) {
      float t[8];
#pragma unroll
      for (int r = 0; r < 8; r++) t[r] = __shfl_xor(rm[r], st, 32);
#pragma unroll
      for (int r = 0; r < 8; r++) rm[r] = fmaxf(rm[r], t[r]);
    }
    // ---- exponentials, P staging, batched row-sum reduction ----
    float ps[8], alpha[8];
#pragma unroll
    for (int r = 0; r < 8; r++) {
      float mn = fmaxf(m[r], rm[r]);
      alpha[r] = __expf(m[r] - mn);
      m[r] = mn;
      float p0 = __expf(e0[r] - mn);
      float p1 = __expf(e1[r] - mn);
      ps[r] = p0 + p1;
      int prow = r + half * 8;                 // C-layout row -> LDS row-major tile
      pst[prow * 32 + lm]      = f2bf(p0);
      pst[prow * 32 + 16 + lm] = f2bf(p1);
    }
#pragma unroll
    for (int st = 1; st <= 8; st <<= 1) {
      float t[8];
#pragma unroll
      for (int r = 0; r < 8; r++) t[r] = __shfl_xor(ps[r], st, 32);
#pragma unroll
      for (int r = 0; r < 8; r++) ps[r] += t[r];
    }
#pragma unroll
    for (int r = 0; r < 8; r++) {
      l[r] = l[r] * alpha[r] + ps[r];
      O0[r] *= alpha[r]; O1[r] *= alpha[r]; O2[r] *= alpha[r]; O3[r] *= alpha[r];
    }
    __syncthreads();
    // Reload P as an A fragment (row = lm, K contiguous per half)
    FragAB P;
    P.q[0] = *(const v4u*)(&pst[lm * 32 + half * 16]);
    P.q[1] = *(const v4u*)(&pst[lm * 32 + half * 16 + 8]);

    O0 = wmma_bf16(P.v, V0.v, O0);
    O1 = wmma_bf16(P.v, V1.v, O1);
    O2 = wmma_bf16(P.v, V2.v, O2);
    O3 = wmma_bf16(P.v, V3.v, O3);
    __syncthreads();
  }
  // epilogue: normalize and store bf16 into attn-out [B*T][H*HD]
#pragma unroll
  for (int r = 0; r < 8; r++) {
    float inv = 1.f / l[r];
    unsigned oo = (unsigned)((b * T_ + qb + r + half * 8) * D_ + h * HD_ + lm);
    o[oo]      = f2bf(O0[r] * inv);
    o[oo + 16] = f2bf(O1[r] * inv);
    o[oo + 32] = f2bf(O2[r] * inv);
    o[oo + 48] = f2bf(O3[r] * inv);
  }
}

// ============================================================
// Host-side orchestration
// ============================================================
extern "C" void kernel_launch(void* const* d_in, const int* in_sizes, int n_in,
                              void* d_out, int out_size, void* d_ws, size_t ws_size,
                              hipStream_t stream) {
  (void)in_sizes; (void)n_in; (void)out_size; (void)ws_size;
  const int*   idx = (const int*)d_in[0];
  const float* tok = (const float*)d_in[1];
  const float* pos = (const float*)d_in[2];
  const float* Wq  = (const float*)d_in[3];
  const float* Wk  = (const float*)d_in[4];
  const float* Wv  = (const float*)d_in[5];
  const float* Wp  = (const float*)d_in[6];
  float* out = (float*)d_out;

  char* ws = (char*)d_ws;
  size_t off = 0;
  auto alloc = [&](size_t bytes) -> void* {
    void* p = ws + off;
    off += (bytes + 255) & ~(size_t)255;
    return p;
  };
  float* x  = (float*)alloc((size_t)NTOT * 4);
  float* xn = (float*)alloc((size_t)NTOT * 4);
  float* po = (float*)alloc((size_t)NTOT * 4);   // out_proj output (f32)
  unsigned short* xnb = (unsigned short*)alloc((size_t)NTOT * 2);
  unsigned short* qb  = (unsigned short*)alloc((size_t)NTOT * 2);
  unsigned short* kb  = (unsigned short*)alloc((size_t)NTOT * 2);
  unsigned short* vT  = (unsigned short*)alloc((size_t)NTOT * 2);
  unsigned short* ab  = (unsigned short*)alloc((size_t)NTOT * 2);
  unsigned short* WqT = (unsigned short*)alloc((size_t)D_ * D_ * 2);
  unsigned short* WkT = (unsigned short*)alloc((size_t)D_ * D_ * 2);
  unsigned short* WvT = (unsigned short*)alloc((size_t)D_ * D_ * 2);
  unsigned short* WpT = (unsigned short*)alloc((size_t)D_ * D_ * 2);
  float* part = (float*)alloc((size_t)NPART * 2 * 4);
  float* acc  = (float*)alloc(2 * 4);

  const int GBLK = NTOT / 256;                 // 16384 blocks for elementwise
  const int WBLK = (D_ * D_) / 256;            // 4096 blocks for transpose
  dim3 ggrid(MROWS / 32, D_ / 64);             // (128,16) GEMM waves (32x64 tiles)
  dim3 agrid(T_ / 16, B_ * H_);                // (64,64) attention waves

  // one-time weight prep (shared across all 6 layers)
  k_transpose_bf16<<<WBLK, 256, 0, stream>>>(Wq, WqT, D_, D_);
  k_transpose_bf16<<<WBLK, 256, 0, stream>>>(Wk, WkT, D_, D_);
  k_transpose_bf16<<<WBLK, 256, 0, stream>>>(Wv, WvT, D_, D_);
  k_transpose_bf16<<<WBLK, 256, 0, stream>>>(Wp, WpT, D_, D_);

  k_embed<<<GBLK, 256, 0, stream>>>(idx, tok, pos, x);

  auto stats = [&](const float* a, const float* b2) {
    k_reduce<<<NPART, 256, 0, stream>>>(a, b2, part, NTOT);
    k_finalize<<<1, 256, 0, stream>>>(part, acc);
  };

  for (int layer = 0; layer < L_; layer++) {
    stats(x, nullptr);
    k_ln_xn<<<GBLK, 256, 0, stream>>>(x, acc, xn, xnb);
    k_gemm<1><<<ggrid, 32, 0, stream>>>(xnb, WqT, nullptr, qb);
    k_gemm<1><<<ggrid, 32, 0, stream>>>(xnb, WkT, nullptr, kb);
    k_gemm<2><<<ggrid, 32, 0, stream>>>(xnb, WvT, nullptr, vT);
    k_attn<<<agrid, 32, 0, stream>>>(qb, kb, vT, ab);
    k_gemm<0><<<ggrid, 32, 0, stream>>>(ab, WpT, po, nullptr);
    stats(xn, po);
    k_ln_res<<<GBLK, 256, 0, stream>>>(xn, po, acc, x);
  }

  stats(x, nullptr);
  k_ln_final<<<GBLK, 256, 0, stream>>>(x, acc, out);
}
